// BCLSTM_54838142435821
// MI455X (gfx1250) — matrix operand order repaired and checked by reading, not compiled, not run on hardware
//
#include <hip/hip_runtime.h>

typedef __attribute__((ext_vector_type(16))) __bf16 v16bf;
typedef __attribute__((ext_vector_type(8)))  float  v8f;
typedef __attribute__((ext_vector_type(4)))  unsigned int u32x4;

#ifndef BB
#define BB 256   // batch
#define TT 128   // timesteps
#endif

// ---------------- helpers ----------------

__device__ __forceinline__ unsigned short f2bf(float f) {
  unsigned int u = __float_as_uint(f);
  u += 0x7FFFu + ((u >> 16) & 1u);          // round-to-nearest-even
  return (unsigned short)(u >> 16);
}
__device__ __forceinline__ float sigf(float x) { return 1.0f / (1.0f + expf(-x)); }

__device__ __forceinline__ v16bf pack_frag(u32x4 lo, u32x4 hi) {
  union { u32x4 q[2]; v16bf v; } u;
  u.q[0] = lo; u.q[1] = hi;
  return u.v;
}

// A fragment (16x32 bf16): row-major source [*, ldk]; lanes 0-15 hold K 0..7,16..23,
// lanes 16-31 hold K 8..15,24..31 (per CDNA5 ISA 16-bit A layout).
__device__ __forceinline__ v16bf frag_a(const unsigned short* A, int ldk,
                                        int mbase, int kbase, int lane) {
  int m  = mbase + (lane & 15);
  int ko = (lane < 16) ? 0 : 8;
  const unsigned short* p = A + (size_t)m * ldk + kbase + ko;
  return pack_frag(*(const u32x4*)p, *(const u32x4*)(p + 16));
}

// B fragment (32x16): we compute A @ W^T with W row-major [N, ldk]; column n of B
// is row n of W. Lanes 0-15 hold K 0..7,16..23, lanes 16-31 hold K 8..15,24..31 (matches A).
__device__ __forceinline__ v16bf frag_b(const unsigned short* W, int ldk,
                                        int nbase, int kbase, int lane) {
  int n  = nbase + (lane & 15);
  int ko = (lane < 16) ? 0 : 8;
  const unsigned short* p = W + (size_t)n * ldk + kbase + ko;
  return pack_frag(*(const u32x4*)p, *(const u32x4*)(p + 16));
}

__device__ __forceinline__ v8f wmma_bf16(v16bf a, v16bf b, v8f c) {
  return __builtin_amdgcn_wmma_f32_16x16x32_bf16(false, a, false, b, (short)0, c,
                                                 false, false);
}

// ---------------- weight fp32 -> bf16 (zero padded) ----------------

__global__ void k_cvt_w(const float* __restrict__ src, unsigned short* __restrict__ dst,
                        int N, int K, int Npad, int Kpad) {
  int idx = blockIdx.x * blockDim.x + threadIdx.x;
  if (idx >= Npad * Kpad) return;
  int n = idx / Kpad, k = idx % Kpad;
  float v = (n < N && k < K) ? src[(size_t)n * K + k] : 0.0f;
  dst[idx] = f2bf(v);
}

// ---------------- masked input [B,T,D] -> bf16 [T*B, Kpad] ----------------

__global__ void k_prep_x(const float* __restrict__ mod, const float* __restrict__ umask,
                         unsigned short* __restrict__ xm, int D, int Kpad) {
  long long idx = (long long)blockIdx.x * blockDim.x + threadIdx.x;
  long long total = (long long)TT * BB * Kpad;
  if (idx >= total) return;
  int k = (int)(idx % Kpad);
  long long row = idx / Kpad;
  int t = (int)(row / BB), b = (int)(row % BB);
  float v = 0.0f;
  if (k < D) v = mod[((size_t)b * TT + t) * D + k] * umask[(size_t)b * TT + t];
  xm[idx] = f2bf(v);
}

// ---------------- generic WMMA GEMM: C[M,N] = A[M,Kpad] @ W[N,Kpad]^T + b0 + b1 ---

__global__ void k_gemm(const unsigned short* __restrict__ A, const unsigned short* __restrict__ W,
                       const float* __restrict__ bias0, const float* __restrict__ bias1,
                       float* __restrict__ C, int M, int N, int Kpad) {
  int lane = threadIdx.x & 31;
  int wave = threadIdx.x >> 5;
  int gw = blockIdx.x * (blockDim.x >> 5) + wave;
  int strips = N >> 6;                      // 64-wide N strips per wave
  int mt = gw / strips;
  int ns = gw % strips;
  if (mt * 16 >= M) return;                 // wave-uniform

  v8f acc[4];
#pragma unroll
  for (int j = 0; j < 4; ++j) acc[j] = (v8f){0,0,0,0,0,0,0,0};

  int kcn = Kpad >> 5;
  for (int kc = 0; kc < kcn; ++kc) {
    v16bf a = frag_a(A, Kpad, mt * 16, kc * 32, lane);
#pragma unroll
    for (int j = 0; j < 4; ++j) {
      v16bf b = frag_b(W, Kpad, ns * 64 + j * 16, kc * 32, lane);
      acc[j] = wmma_bf16(a, b, acc[j]);
    }
  }
  int col = (lane & 15);
  int r0  = mt * 16 + ((lane < 16) ? 0 : 8);
#pragma unroll
  for (int j = 0; j < 4; ++j) {
    int n = ns * 64 + j * 16 + col;
    float bs = (bias0 ? bias0[n] : 0.0f) + (bias1 ? bias1[n] : 0.0f);
#pragma unroll
    for (int v = 0; v < 8; ++v)
      C[(size_t)(r0 + v) * N + n] = acc[j][v] + bs;
  }
}

// ---------------- persistent LSTM scan (one workgroup, 32 waves) ----------------
// Gin: [T, 256, 4H] fp32 (input proj + bias).  Whh: bf16 [4H, H].  Wfc: bf16 [FPAD, H].
// MODALITY: extra tanh on h; output = tanh(h@fc^T+b)*umask as bf16 into dfeat[T*256,256].
// else: output = tanh(h@fc^T+b) fp32 into outF[T*256, FPAD].

template <int H, int FPAD, bool MODALITY>
__global__ __launch_bounds__(1024) void k_scan(
    const float* __restrict__ Gin, const unsigned short* __restrict__ Whh,
    const unsigned short* __restrict__ Wfc, const float* __restrict__ bfc,
    const float* __restrict__ umask, unsigned short* __restrict__ outBF,
    float* __restrict__ outF, int colOff, int F) {
  extern __shared__ unsigned short h_sh[];   // [256 * H] bf16 (tanh'd hidden state)
  constexpr int TPW = H / 32;                // 16*(H/16) tiles over 32 waves
  constexpr int KC  = H / 32;
  int lane = threadIdx.x & 31, wave = threadIdx.x >> 5;
  int col  = lane & 15;
  int rofs = (lane < 16) ? 0 : 8;

  for (int i = threadIdx.x; i < 256 * H; i += 1024) h_sh[i] = 0;
  float c_reg[TPW][8];
#pragma unroll
  for (int p = 0; p < TPW; ++p)
#pragma unroll
    for (int v = 0; v < 8; ++v) c_reg[p][v] = 0.0f;
  __syncthreads();

  for (int t = 0; t < TT; ++t) {
    float h2loc[TPW][8];
#pragma unroll
    for (int p = 0; p < TPW; ++p) {
      int tile = wave * TPW + p;
      int mt = tile & 15, nt = tile >> 4;
      v8f acc[4];
      const float* gbase = Gin + ((size_t)t * 256 + mt * 16 + rofs) * (4 * H);
#pragma unroll
      for (int g = 0; g < 4; ++g) {
        int n = g * H + nt * 16 + col;
#pragma unroll
        for (int v = 0; v < 8; ++v) acc[g][v] = gbase[(size_t)v * (4 * H) + n];
      }
#pragma unroll
      for (int kc = 0; kc < KC; ++kc) {
        v16bf a = frag_a(h_sh, H, mt * 16, kc * 32, lane);
#pragma unroll
        for (int g = 0; g < 4; ++g) {
          v16bf b = frag_b(Whh, H, g * H + nt * 16, kc * 32, lane);
          acc[g] = wmma_bf16(a, b, acc[g]);
        }
      }
#pragma unroll
      for (int v = 0; v < 8; ++v) {
        float i_ = sigf(acc[0][v]);
        float f_ = sigf(acc[1][v]);
        float g_ = tanhf(acc[2][v]);
        float o_ = sigf(acc[3][v]);
        float c2 = f_ * c_reg[p][v] + i_ * g_;
        c_reg[p][v] = c2;
        float h2 = o_ * tanhf(c2);
        if (MODALITY) h2 = tanhf(h2);        // reference applies extra tanh to carried h
        h2loc[p][v] = h2;
      }
    }
    __syncthreads();                          // all reads of old h done
#pragma unroll
    for (int p = 0; p < TPW; ++p) {
      int tile = wave * TPW + p;
      int mt = tile & 15, nt = tile >> 4;
#pragma unroll
      for (int v = 0; v < 8; ++v)
        h_sh[(size_t)(mt * 16 + rofs + v) * H + nt * 16 + col] = f2bf(h2loc[p][v]);
    }
    __syncthreads();                          // new h visible to all waves

    // per-step FC: y = tanh(h2 @ Wfc^T + bfc)
    for (int tile = wave; tile < FPAD; tile += 32) {   // 16*(FPAD/16) tiles
      int mt = tile & 15, nt = tile >> 4;
      int n = nt * 16 + col;
      float bs = (n < F) ? bfc[n] : 0.0f;
      v8f acc = (v8f){0,0,0,0,0,0,0,0};
#pragma unroll
      for (int v = 0; v < 8; ++v) acc[v] = bs;
#pragma unroll
      for (int kc = 0; kc < KC; ++kc) {
        v16bf a = frag_a(h_sh, H, mt * 16, kc * 32, lane);
        v16bf b = frag_b(Wfc, H, nt * 16, kc * 32, lane);
        acc = wmma_bf16(a, b, acc);
      }
      int r0 = mt * 16 + rofs;
      if (MODALITY) {
#pragma unroll
        for (int v = 0; v < 8; ++v) {
          if (n < F) {
            float um = umask[(size_t)(r0 + v) * TT + t];
            outBF[((size_t)t * 256 + r0 + v) * 256 + colOff + n] = f2bf(tanhf(acc[v]) * um);
          }
        }
      } else {
#pragma unroll
        for (int v = 0; v < 8; ++v)
          outF[((size_t)t * 256 + r0 + v) * FPAD + n] = tanhf(acc[v]);
      }
    }
    // no barrier needed here: next step only reads h_sh until its own barrier
  }
}

// ---------------- output head: logits + log_softmax ----------------

__global__ void k_head(const float* __restrict__ hs, const float* __restrict__ sw,
                       const float* __restrict__ sb, float* __restrict__ out) {
  int idx = blockIdx.x * blockDim.x + threadIdx.x;
  if (idx >= BB * TT) return;
  int t = idx / BB, b = idx % BB;
  const float* h = hs + (size_t)idx * 128;
  float l[6];
#pragma unroll
  for (int c = 0; c < 6; ++c) {
    float s = sb[c];
    const float* w = sw + c * 128;
    for (int k = 0; k < 128; ++k) s += h[k] * w[k];
    l[c] = s;
  }
  float m = l[0];
#pragma unroll
  for (int c = 1; c < 6; ++c) m = fmaxf(m, l[c]);
  float s = 0.0f;
#pragma unroll
  for (int c = 0; c < 6; ++c) s += expf(l[c] - m);
  float ls = m + logf(s);
  float* o = out + ((size_t)b * TT + t) * 6;
#pragma unroll
  for (int c = 0; c < 6; ++c) o[c] = l[c] - ls;
}

// ---------------- workspace layout ----------------

constexpr size_t OFF_WIHP0 = 0;
constexpr size_t OFF_WIHP1 = OFF_WIHP0 + (size_t)512 * 320 * 2;
constexpr size_t OFF_WIHP2 = OFF_WIHP1 + (size_t)256 * 128 * 2;
constexpr size_t OFF_WHH0  = OFF_WIHP2 + (size_t)512 * 512 * 2;
constexpr size_t OFF_WHH1  = OFF_WHH0  + (size_t)512 * 128 * 2;
constexpr size_t OFF_WHH2  = OFF_WHH1  + (size_t)256 * 64 * 2;
constexpr size_t OFF_FCW0  = OFF_WHH2  + (size_t)512 * 128 * 2;
constexpr size_t OFF_FCW1  = OFF_FCW0  + (size_t)112 * 128 * 2;
constexpr size_t OFF_FCW2  = OFF_FCW1  + (size_t)64 * 64 * 2;
constexpr size_t OFF_WIHD  = OFF_FCW2  + (size_t)112 * 128 * 2;
constexpr size_t OFF_WHHD  = OFF_WIHD  + (size_t)1024 * 256 * 2;
constexpr size_t OFF_FCOW  = OFF_WHHD  + (size_t)1024 * 256 * 2;
constexpr size_t OFF_XM    = ((OFF_FCOW + (size_t)128 * 256 * 2) + 255) & ~(size_t)255;
constexpr size_t OFF_GIN0  = OFF_XM   + (size_t)32768 * 512 * 2;
constexpr size_t OFF_GIN1  = OFF_GIN0 + (size_t)32768 * 512 * 4;
constexpr size_t OFF_GIN2  = OFF_GIN1 + (size_t)32768 * 256 * 4;
constexpr size_t OFF_DFEAT = OFF_GIN2 + (size_t)32768 * 512 * 4;
constexpr size_t OFF_GD    = OFF_XM;     // aliases xm+gin0+gin1 (exactly 128 MiB, dead)
constexpr size_t OFF_HS    = OFF_GIN2;   // aliases gin2 (dead)
constexpr size_t TOTAL_WS  = OFF_DFEAT + (size_t)32768 * 256 * 2;

extern "C" void kernel_launch(void* const* d_in, const int* in_sizes, int n_in,
                              void* d_out, int out_size, void* d_ws, size_t ws_size,
                              hipStream_t stream) {
  (void)in_sizes; (void)n_in; (void)out_size;
  if (ws_size < TOTAL_WS) return;

  const float* mod[3]  = {(const float*)d_in[0], (const float*)d_in[1], (const float*)d_in[2]};
  const float* umask   = (const float*)d_in[3];
  const float* w_ih[3] = {(const float*)d_in[4],  (const float*)d_in[10], (const float*)d_in[16]};
  const float* w_hh[3] = {(const float*)d_in[5],  (const float*)d_in[11], (const float*)d_in[17]};
  const float* b_ih[3] = {(const float*)d_in[6],  (const float*)d_in[12], (const float*)d_in[18]};
  const float* b_hh[3] = {(const float*)d_in[7],  (const float*)d_in[13], (const float*)d_in[19]};
  const float* fc_w[3] = {(const float*)d_in[8],  (const float*)d_in[14], (const float*)d_in[20]};
  const float* fc_b[3] = {(const float*)d_in[9],  (const float*)d_in[15], (const float*)d_in[21]};
  const float* w_ih_d  = (const float*)d_in[22];
  const float* w_hh_d  = (const float*)d_in[23];
  const float* b_ih_d  = (const float*)d_in[24];
  const float* b_hh_d  = (const float*)d_in[25];
  const float* fco_w   = (const float*)d_in[26];
  const float* fco_b   = (const float*)d_in[27];
  const float* smax_w  = (const float*)d_in[28];
  const float* smax_b  = (const float*)d_in[29];
  float* out = (float*)d_out;
  char*  ws  = (char*)d_ws;

  auto bf = [&](size_t off) { return (unsigned short*)(ws + off); };
  auto fp = [&](size_t off) { return (float*)(ws + off); };
  auto cvt = [&](const float* src, size_t off, int N, int K, int Np, int Kp) {
    int total = Np * Kp;
    k_cvt_w<<<(total + 255) / 256, 256, 0, stream>>>(src, bf(off), N, K, Np, Kp);
  };

  // allow >64KB dynamic LDS on the persistent scan kernels
  hipFuncSetAttribute(reinterpret_cast<const void*>(&k_scan<128, 112, true>),
                      hipFuncAttributeMaxDynamicSharedMemorySize, 65536);
  hipFuncSetAttribute(reinterpret_cast<const void*>(&k_scan<64, 64, true>),
                      hipFuncAttributeMaxDynamicSharedMemorySize, 32768);
  hipFuncSetAttribute(reinterpret_cast<const void*>(&k_scan<256, 128, false>),
                      hipFuncAttributeMaxDynamicSharedMemorySize, 131072);

  // -- weights -> bf16 (zero padded) --
  cvt(w_ih[0], OFF_WIHP0, 512, 300, 512, 320);
  cvt(w_ih[1], OFF_WIHP1, 256, 100, 256, 128);
  cvt(w_ih[2], OFF_WIHP2, 512, 512, 512, 512);
  cvt(w_hh[0], OFF_WHH0,  512, 128, 512, 128);
  cvt(w_hh[1], OFF_WHH1,  256, 64,  256, 64);
  cvt(w_hh[2], OFF_WHH2,  512, 128, 512, 128);
  cvt(fc_w[0], OFF_FCW0,  100, 128, 112, 128);
  cvt(fc_w[1], OFF_FCW1,  50,  64,  64,  64);
  cvt(fc_w[2], OFF_FCW2,  100, 128, 112, 128);
  cvt(w_ih_d,  OFF_WIHD,  1024, 250, 1024, 256);
  cvt(w_hh_d,  OFF_WHHD,  1024, 256, 1024, 256);
  cvt(fco_w,   OFF_FCOW,  128, 256, 128, 256);

  hipMemsetAsync(ws + OFF_DFEAT, 0, (size_t)32768 * 256 * 2, stream);

  const int M = TT * BB;  // 32768
  const int din[3]  = {300, 100, 512};
  const int kpad[3] = {320, 128, 512};
  const int n4h[3]  = {512, 256, 512};
  const size_t wihp[3] = {OFF_WIHP0, OFF_WIHP1, OFF_WIHP2};
  const size_t ginoff[3] = {OFF_GIN0, OFF_GIN1, OFF_GIN2};

  // -- batched input projections: Gin[i] = (x*umask) @ Wih^T + (b_ih + b_hh) --
  for (int i = 0; i < 3; ++i) {
    long long total = (long long)M * kpad[i];
    k_prep_x<<<(unsigned)((total + 255) / 256), 256, 0, stream>>>(mod[i], umask,
                                                                  bf(OFF_XM), din[i], kpad[i]);
    int waves = (M / 16) * (n4h[i] / 64);
    k_gemm<<<waves / 8, 256, 0, stream>>>(bf(OFF_XM), bf(wihp[i]), b_ih[i], b_hh[i],
                                          fp(ginoff[i]), M, n4h[i], kpad[i]);
    // -- persistent recurrent scan for this modality (must run before XM is reused) --
    if (i == 0)
      k_scan<128, 112, true><<<1, 1024, 65536, stream>>>(fp(OFF_GIN0), bf(OFF_WHH0),
          bf(OFF_FCW0), fc_b[0], umask, bf(OFF_DFEAT), nullptr, 0, 100);
    else if (i == 1)
      k_scan<64, 64, true><<<1, 1024, 32768, stream>>>(fp(OFF_GIN1), bf(OFF_WHH1),
          bf(OFF_FCW1), fc_b[1], umask, bf(OFF_DFEAT), nullptr, 100, 50);
    else
      k_scan<128, 112, true><<<1, 1024, 65536, stream>>>(fp(OFF_GIN2), bf(OFF_WHH2),
          bf(OFF_FCW2), fc_b[2], umask, bf(OFF_DFEAT), nullptr, 150, 100);
  }

  // -- dialogue-level input projection: Gd = dfeat @ Wihd^T + (b_ih_d + b_hh_d) --
  {
    int waves = (M / 16) * (1024 / 64);
    k_gemm<<<waves / 8, 256, 0, stream>>>(bf(OFF_DFEAT), bf(OFF_WIHD), b_ih_d, b_hh_d,
                                          fp(OFF_GD), M, 1024, 256);
  }
  // -- dialogue recurrent scan -> hs = tanh(h @ fc_out^T + b) --
  k_scan<256, 128, false><<<1, 1024, 131072, stream>>>(fp(OFF_GD), bf(OFF_WHHD),
      bf(OFF_FCOW), fco_b, umask, nullptr, fp(OFF_HS), 0, 128);

  // -- logits + log_softmax --
  k_head<<<(BB * TT) / 256, 256, 0, stream>>>(fp(OFF_HS), smax_w, smax_b, out);
}
